// GRU_18562848653428
// MI455X (gfx1250) — matrix operand order repaired and checked by reading, not compiled, not run on hardware
//
#include <hip/hip_runtime.h>

#define DEVINL __device__ __forceinline__

typedef __bf16 bf16;
typedef __attribute__((ext_vector_type(16))) bf16  v16bf;
typedef __attribute__((ext_vector_type(8)))  float v8f;

static constexpr int BSZ = 64;     // batch
static constexpr int SEQ = 1024;   // sequence length
static constexpr int NIN = 256;    // input
static constexpr int HID = 512;    // hidden
static constexpr int NOUT = 256;   // output
static constexpr unsigned RECUR_BLOCKS = HID / 64;   // 8 persistent blocks

union FragBF { v16bf v; uint4 q[2]; };

DEVINL bf16 f2bf(float f) {
  union { float f; unsigned u; } a; a.f = f;
  unsigned u = a.u;
  u += 0x7FFFu + ((u >> 16) & 1u);            // round-to-nearest-even
  unsigned short h = (unsigned short)(u >> 16);
  bf16 r; __builtin_memcpy(&r, &h, sizeof(r));
  return r;
}

DEVINL float sigmoidf_(float x) { return 1.0f / (1.0f + __expf(-x)); }

// A-matrix 16x32 bf16 fragment (ISA 7.12.2): lane L holds row M=L&15;
// elements 0..7 = K kb..kb+7, elements 8..15 = K kb+16..kb+23, kb=(L>>4)*8.
DEVINL v16bf load_A(const bf16* __restrict__ base, long row_stride, int lane) {
  const bf16* p = base + (long)(lane & 15) * row_stride + ((lane >> 4) << 3);
  FragBF f;
  f.q[0] = *(const uint4*)(p);
  f.q[1] = *(const uint4*)(p + 16);
  return f.v;
}

// B-matrix 32x16 bf16 fragment: lane L holds col N=L&15; elements 0..7 =
// K kb..kb+7, elements 8..15 = K kb+16..kb+23, kb=(L>>4)*8. Weights stored transposed WT[N][K].
DEVINL v16bf load_B(const bf16* __restrict__ base, long row_stride, int lane) {
  const bf16* p = base + (long)(lane & 15) * row_stride + ((lane >> 4) << 3);
  FragBF f;
  f.q[0] = *(const uint4*)(p);
  f.q[1] = *(const uint4*)(p + 16);
  return f.v;
}

DEVINL v8f wmma_bf16(v16bf a, v16bf b, v8f c) {
  return __builtin_amdgcn_wmma_f32_16x16x32_bf16(false, a, false, b, (short)0, c,
                                                 false, false);
}

// Device-scope sense-reversing grid barrier (8 co-resident blocks).
// Acquire loads at AGENT scope emit the gfx1250 device-scope cache
// invalidate so post-barrier reads see other WGPs' stores.
DEVINL void grid_barrier(unsigned* __restrict__ count, unsigned* __restrict__ sense,
                         unsigned nblk) {
  __syncthreads();
  if (threadIdx.x == 0) {
    const unsigned old =
        __hip_atomic_load(sense, __ATOMIC_RELAXED, __HIP_MEMORY_SCOPE_AGENT);
    const unsigned arrived =
        __hip_atomic_fetch_add(count, 1u, __ATOMIC_ACQ_REL, __HIP_MEMORY_SCOPE_AGENT);
    if (arrived == nblk - 1u) {
      __hip_atomic_store(count, 0u, __ATOMIC_RELAXED, __HIP_MEMORY_SCOPE_AGENT);
      __hip_atomic_fetch_add(sense, 1u, __ATOMIC_RELEASE, __HIP_MEMORY_SCOPE_AGENT);
    } else {
      while (__hip_atomic_load(sense, __ATOMIC_ACQUIRE, __HIP_MEMORY_SCOPE_AGENT) == old) {
        __builtin_amdgcn_s_sleep(8);
      }
    }
  }
  __syncthreads();
  __threadfence();
}

// ---------------------------------------------------------------- prep kernels
__global__ void __launch_bounds__(256) cvt_bf16_kernel(const float* __restrict__ in,
                                                       bf16* __restrict__ out, long n) {
  long i = (long)blockIdx.x * blockDim.x + threadIdx.x;
  if (i < n) out[i] = f2bf(in[i]);
}

// W[K][N] f32 -> WT[N][K] bf16
__global__ void __launch_bounds__(256) transpose_cvt_kernel(const float* __restrict__ W,
                                                            bf16* __restrict__ WT,
                                                            int K, int N) {
  long i = (long)blockIdx.x * blockDim.x + threadIdx.x;
  long total = (long)K * N;
  if (i >= total) return;
  int nn = (int)(i / K);
  int kk = (int)(i % K);
  WT[i] = f2bf(W[(long)kk * N + nn]);
}

__global__ void __launch_bounds__(256) zero_state_kernel(float* __restrict__ Hf,
                                                         bf16* __restrict__ Hbf,
                                                         unsigned* __restrict__ bar) {
  int i = blockIdx.x * blockDim.x + threadIdx.x;
  if (i < BSZ * HID) { Hf[i] = 0.0f; Hbf[i] = f2bf(0.0f); }
  if (i < 2) bar[i] = 0u;   // barrier {count, sense}
}

__global__ void __launch_bounds__(256) copy_ht_kernel(const float* __restrict__ Hf,
                                                      float* __restrict__ out) {
  int i = blockIdx.x * blockDim.x + threadIdx.x;
  if (i < BSZ * HID) out[i] = Hf[i];
}

// ------------------------------------------------- persistent recurrence kernel
// One launch for all SEQ steps. 8 blocks x 128 threads (4 waves); block b owns
// hidden columns [64b, 64b+64) in both phases; wave = M-tile (batch rows 16w..).
// Phase 1: R = sigmoid(X_t@Wxr + H@Whr + br); Z = sigmoid(X_t@Wxz + H@Whz + bz)
//          -> Z (f32, block-local), RH = R*H (bf16, crosses blocks)
// Phase 2: Htil = tanh(X_t@Wxh + RH@Whh + bh); Hnew = Z*H + (1-Z)*Htil
//          -> Hf/Hbf (state), Hs[b*SEQ+t] (history for output GEMM)
__global__ void __launch_bounds__(128) gru_recur_kernel(
    const bf16*  __restrict__ Xbf,    // [BSZ,SEQ,NIN]
    const bf16*  __restrict__ WhrT,   // [HID,HID]
    const bf16*  __restrict__ WhzT,   // [HID,HID]
    const bf16*  __restrict__ WhhT,   // [HID,HID]
    const bf16*  __restrict__ WxrT,   // [HID,NIN]
    const bf16*  __restrict__ WxzT,   // [HID,NIN]
    const bf16*  __restrict__ WxhT,   // [HID,NIN]
    const float* __restrict__ br,     // [HID]
    const float* __restrict__ bz,     // [HID]
    const float* __restrict__ bh,     // [HID]
    float*       __restrict__ Hf,     // [BSZ,HID] (in/out)
    bf16*        __restrict__ Hbf,    // [BSZ,HID] (in/out)
    bf16*        __restrict__ RHbf,   // [BSZ,HID]
    float*       __restrict__ Z,      // [BSZ,HID]
    bf16*        __restrict__ Hs,     // [BSZ*SEQ,HID], row = b*SEQ+t
    unsigned*    __restrict__ bar) {  // [2] = {count, sense}
  const int lane  = threadIdx.x & 31;
  const int wave  = threadIdx.x >> 5;     // 0..3 -> M-tile
  const int m0    = wave * 16;
  const int nbase = blockIdx.x * 64;
  const int hi    = (lane >> 4) << 3;

  for (int t = 0; t < SEQ; t++) {
    // ---------------- phase 1: R and Z gates ----------------
    {
      v8f accR[4] = {}, accZ[4] = {};
      for (int k = 0; k < HID; k += 32) {
        v16bf a = load_A(Hbf + (long)m0 * HID + k, HID, lane);
#pragma unroll
        for (int nt = 0; nt < 4; nt++) {
          const long n0 = nbase + nt * 16;
          v16bf b0 = load_B(WhrT + n0 * HID + k, HID, lane);
          v16bf b1 = load_B(WhzT + n0 * HID + k, HID, lane);
          accR[nt] = wmma_bf16(a, b0, accR[nt]);
          accZ[nt] = wmma_bf16(a, b1, accZ[nt]);
        }
      }
      for (int k = 0; k < NIN; k += 32) {
        v16bf a = load_A(Xbf + ((long)m0 * SEQ + t) * NIN + k, (long)SEQ * NIN, lane);
#pragma unroll
        for (int nt = 0; nt < 4; nt++) {
          const long n0 = nbase + nt * 16;
          v16bf b0 = load_B(WxrT + n0 * NIN + k, NIN, lane);
          v16bf b1 = load_B(WxzT + n0 * NIN + k, NIN, lane);
          accR[nt] = wmma_bf16(a, b0, accR[nt]);
          accZ[nt] = wmma_bf16(a, b1, accZ[nt]);
        }
      }
#pragma unroll
      for (int nt = 0; nt < 4; nt++) {
        const int n = nbase + nt * 16 + (lane & 15);
        const float brv = br[n], bzv = bz[n];
#pragma unroll
        for (int r = 0; r < 8; r++) {
          const int m = m0 + r + hi;
          const long idx = (long)m * HID + n;
          const float rv = sigmoidf_(accR[nt][r] + brv);
          const float zv = sigmoidf_(accZ[nt][r] + bzv);
          Z[idx]    = zv;
          RHbf[idx] = f2bf(rv * Hf[idx]);
        }
      }
    }
    grid_barrier(&bar[0], &bar[1], RECUR_BLOCKS);

    // ---------------- phase 2: candidate + state update ----------------
    {
      v8f acc[4] = {};
      for (int k = 0; k < HID; k += 32) {
        v16bf a = load_A(RHbf + (long)m0 * HID + k, HID, lane);
#pragma unroll
        for (int nt = 0; nt < 4; nt++) {
          const long n0 = nbase + nt * 16;
          acc[nt] = wmma_bf16(a, load_B(WhhT + n0 * HID + k, HID, lane), acc[nt]);
        }
      }
      for (int k = 0; k < NIN; k += 32) {
        v16bf a = load_A(Xbf + ((long)m0 * SEQ + t) * NIN + k, (long)SEQ * NIN, lane);
#pragma unroll
        for (int nt = 0; nt < 4; nt++) {
          const long n0 = nbase + nt * 16;
          acc[nt] = wmma_bf16(a, load_B(WxhT + n0 * NIN + k, NIN, lane), acc[nt]);
        }
      }
#pragma unroll
      for (int nt = 0; nt < 4; nt++) {
        const int n = nbase + nt * 16 + (lane & 15);
        const float bhv = bh[n];
#pragma unroll
        for (int r = 0; r < 8; r++) {
          const int m = m0 + r + hi;
          const long idx = (long)m * HID + n;
          const float htil = tanhf(acc[nt][r] + bhv);
          const float zv   = Z[idx];
          const float hnew = zv * Hf[idx] + (1.0f - zv) * htil;
          Hf[idx]  = hnew;
          const bf16 hb = f2bf(hnew);
          Hbf[idx] = hb;
          Hs[((long)m * SEQ + t) * HID + n] = hb;
        }
      }
    }
    grid_barrier(&bar[0], &bar[1], RECUR_BLOCKS);
  }
}

// --------------------------------------------------------------- output GEMM
// out[b,s,:] = Hs[b*SEQ+s,:] @ Whq + bq ; M = BSZ*SEQ, K = HID, N = NOUT
// grid = (M/64, N/64), block = 128 (4 waves, one M-tile each)
__global__ void __launch_bounds__(128) out_gemm_kernel(
    const bf16*  __restrict__ Hs,     // [BSZ*SEQ,HID]
    const bf16*  __restrict__ WhqT,   // [NOUT,HID]
    const float* __restrict__ bq,     // [NOUT]
    float*       __restrict__ out) {  // [BSZ*SEQ,NOUT]
  const int lane  = threadIdx.x & 31;
  const int wave  = threadIdx.x >> 5;
  const long m0   = (long)blockIdx.x * 64 + wave * 16;
  const int nbase = blockIdx.y * 64;

  v8f acc[4] = {};
  for (int k = 0; k < HID; k += 32) {
    v16bf a = load_A(Hs + m0 * HID + k, HID, lane);
#pragma unroll
    for (int nt = 0; nt < 4; nt++) {
      const long n0 = nbase + nt * 16;
      acc[nt] = wmma_bf16(a, load_B(WhqT + n0 * HID + k, HID, lane), acc[nt]);
    }
  }

  const int hi = (lane >> 4) << 3;
#pragma unroll
  for (int nt = 0; nt < 4; nt++) {
    const int n = nbase + nt * 16 + (lane & 15);
    const float bqv = bq[n];
#pragma unroll
    for (int r = 0; r < 8; r++) {
      const long m = m0 + r + hi;
      out[m * NOUT + n] = acc[nt][r] + bqv;
    }
  }
}

// ------------------------------------------------------------------- launcher
extern "C" void kernel_launch(void* const* d_in, const int* in_sizes, int n_in,
                              void* d_out, int out_size, void* d_ws, size_t ws_size,
                              hipStream_t stream) {
  (void)in_sizes; (void)n_in; (void)out_size; (void)ws_size;

  const float* X   = (const float*)d_in[0];
  const float* Wxr = (const float*)d_in[1];
  const float* Whr = (const float*)d_in[2];
  const float* br  = (const float*)d_in[3];
  const float* Wxz = (const float*)d_in[4];
  const float* Whz = (const float*)d_in[5];
  const float* bz  = (const float*)d_in[6];
  const float* Wxh = (const float*)d_in[7];
  const float* Whh = (const float*)d_in[8];
  const float* bh  = (const float*)d_in[9];
  const float* Whq = (const float*)d_in[10];
  const float* bq  = (const float*)d_in[11];
  float* out = (float*)d_out;

  // Workspace carve-out (~104 MB total)
  char* w = (char*)d_ws;
  auto carve = [&](size_t bytes) -> void* {
    void* p = (void*)w;
    w += (bytes + 255) & ~(size_t)255;
    return p;
  };
  bf16*     Xbf  = (bf16*)    carve((size_t)BSZ * SEQ * NIN * 2);  // 32 MB
  bf16*     WxrT = (bf16*)    carve((size_t)HID * NIN * 2);
  bf16*     WxzT = (bf16*)    carve((size_t)HID * NIN * 2);
  bf16*     WxhT = (bf16*)    carve((size_t)HID * NIN * 2);
  bf16*     WhrT = (bf16*)    carve((size_t)HID * HID * 2);
  bf16*     WhzT = (bf16*)    carve((size_t)HID * HID * 2);
  bf16*     WhhT = (bf16*)    carve((size_t)HID * HID * 2);
  bf16*     WhqT = (bf16*)    carve((size_t)NOUT * HID * 2);
  float*    Hf   = (float*)   carve((size_t)BSZ * HID * 4);
  bf16*     Hbf  = (bf16*)    carve((size_t)BSZ * HID * 2);
  bf16*     RHbf = (bf16*)    carve((size_t)BSZ * HID * 2);
  float*    Zbuf = (float*)   carve((size_t)BSZ * HID * 4);
  unsigned* bar  = (unsigned*)carve(2 * sizeof(unsigned));
  bf16*     Hs   = (bf16*)    carve((size_t)BSZ * SEQ * HID * 2);  // 64 MB

  // 1) one-time conversions / transposes
  {
    long n = (long)BSZ * SEQ * NIN;
    cvt_bf16_kernel<<<(unsigned)((n + 255) / 256), 256, 0, stream>>>(X, Xbf, n);
  }
  auto tconv = [&](const float* src, bf16* dst, int K, int N) {
    long total = (long)K * N;
    transpose_cvt_kernel<<<(unsigned)((total + 255) / 256), 256, 0, stream>>>(src, dst, K, N);
  };
  tconv(Wxr, WxrT, NIN, HID);
  tconv(Wxz, WxzT, NIN, HID);
  tconv(Wxh, WxhT, NIN, HID);
  tconv(Whr, WhrT, HID, HID);
  tconv(Whz, WhzT, HID, HID);
  tconv(Whh, WhhT, HID, HID);
  tconv(Whq, WhqT, HID, NOUT);

  zero_state_kernel<<<(BSZ * HID + 255) / 256, 256, 0, stream>>>(Hf, Hbf, bar);

  // 2) entire recurrence in ONE persistent kernel (device-scope barriers)
  gru_recur_kernel<<<RECUR_BLOCKS, 128, 0, stream>>>(
      Xbf, WhrT, WhzT, WhhT, WxrT, WxzT, WxhT, br, bz, bh,
      Hf, Hbf, RHbf, Zbuf, Hs, bar);

  // 3) big output GEMM + final-state copy
  {
    dim3 grid((unsigned)((long)BSZ * SEQ / 64), NOUT / 64);
    out_gemm_kernel<<<grid, 128, 0, stream>>>(Hs, WhqT, bq, out);
  }
  copy_ht_kernel<<<(BSZ * HID + 255) / 256, 256, 0, stream>>>(
      Hf, out + (size_t)BSZ * SEQ * NOUT);
}
